// RNNModel_58849641890649
// MI455X (gfx1250) — matrix operand order repaired and checked by reading, not compiled, not run on hardware
//
#include <hip/hip_runtime.h>
#include <hip/hip_bf16.h>

// Problem constants (from reference)
#define B_TOT   4096
#define T_LEN   256
#define I_SZ    272     // input size
#define H_SZ    50      // hidden size
#define KP      288     // K padded to 9*32 for 16x16x32 WMMA
#define NKK     9       // KP/32
#define HP      64      // hidden padded to 4 * 16
#define BM      16      // batch rows per workgroup (one M tile)
#define NTHREADS 128    // 4 wave32s; wave w owns hidden columns [16w, 16w+16)
#define NQ4     (BM * (I_SZ / 4))   // 1088 float4s per x tile = 8*128 + 64
#define NSLOT   8                   // full per-thread staging slots
#define TAILQ   (NQ4 - NSLOT * NTHREADS)   // 64: tail slot valid for tid < 64
#define NPFL    (BM * 9)            // 144 prefetch lines per tile

typedef __attribute__((ext_vector_type(16))) __bf16 v16bf;
typedef __attribute__((ext_vector_type(8)))  __bf16 v8bf;
typedef __attribute__((ext_vector_type(8)))  float  v8f;

// ---- pack two f32 -> two bf16 (truncation) : one v_perm_b32 ----
#if __has_builtin(__builtin_amdgcn_perm)
static __device__ __forceinline__ unsigned pack2(float a, float b) {
    return __builtin_amdgcn_perm(__builtin_bit_cast(unsigned, b),
                                 __builtin_bit_cast(unsigned, a), 0x07060302u);
}
#else
static __device__ __forceinline__ unsigned pack2(float a, float b) {
    return (__builtin_bit_cast(unsigned, b) & 0xffff0000u) |
           (__builtin_bit_cast(unsigned, a) >> 16);
}
#endif

static __device__ __forceinline__ unsigned short bf_hi(float f) {
    return (unsigned short)(__builtin_bit_cast(unsigned, f) >> 16);  // ds_store_b16_d16_hi
}

// ---- tanh: native v_tanh_f32 if present, else branchless exp2/rcp form ----
#if __has_builtin(__builtin_amdgcn_tanhf)
#define TANH_F32(x) __builtin_amdgcn_tanhf(x)
#elif __has_builtin(__builtin_amdgcn_tanh_f32)
#define TANH_F32(x) __builtin_amdgcn_tanh_f32(x)
#else
static __device__ __forceinline__ float TANH_F32(float x) {
    float ax = __builtin_fabsf(x);
    float e  = __builtin_amdgcn_exp2f(ax * -2.885390081777927f); // e^{-2|x|}
    float r  = (1.0f - e) * __builtin_amdgcn_rcpf(1.0f + e);
    return __builtin_copysignf(r, x);
}
#endif

// A fragment (16xK tile, K-chunk of 32) per ISA 7.12.2
static __device__ __forceinline__ v16bf load_A32(const unsigned short* base, int stride,
                                                 int k0, int m, int hi) {
    const __bf16* p = (const __bf16*)(base + m * stride + k0 + hi * 8);
    v8bf lo = *(const v8bf*)p;
    v8bf hh = *(const v8bf*)(p + 16);
    return __builtin_shufflevector(lo, hh, 0,1,2,3,4,5,6,7,8,9,10,11,12,13,14,15);
}

// B fragment (Kx16 tile, K-chunk of 32) from the [n][k] LDS weight copy
static __device__ __forceinline__ v16bf load_B32(const unsigned short* base, int stride,
                                                 int k0, int n, int hi) {
    const __bf16* p = (const __bf16*)(base + n * stride + k0 + hi * 8);
    v8bf lo = *(const v8bf*)p;
    v8bf hh = *(const v8bf*)(p + 16);
    return __builtin_shufflevector(lo, hh, 0,1,2,3,4,5,6,7,8,9,10,11,12,13,14,15);
}

__global__ __launch_bounds__(NTHREADS)
void rnn_fused_kernel(const float* __restrict__ x,    const float* __restrict__ W_ih,
                      const float* __restrict__ W_hh, const float* __restrict__ b_ih,
                      const float* __restrict__ b_hh, const float* __restrict__ W_fc,
                      const float* __restrict__ b_fc, float* __restrict__ out)
{
    __shared__ __attribute__((aligned(32))) unsigned short Wihb[HP * KP];    // [n][k] bf16
    __shared__ __attribute__((aligned(32))) unsigned short Whhb[HP * HP];    // [n][k] bf16
    __shared__ __attribute__((aligned(32))) unsigned short Xb[2][BM * KP];   // [m][k] bf16
    __shared__ __attribute__((aligned(32))) unsigned short Hb[2][BM * HP];   // [m][n] bf16
    __shared__ float biasl[HP];
    __shared__ float Wfcl[HP];
    __shared__ float Hf[BM * HP];

    const int tid  = threadIdx.x;
    const int lane = tid & 31;
    const int wv   = tid >> 5;       // wave id = hidden N-tile
    const int m16  = lane & 15;
    const int hi   = lane >> 4;
    const int b0   = blockIdx.x * BM;

    // ---- zero padded LDS regions (pads stay zero; h0 = 0) ----
    for (int i = tid; i < HP * KP; i += NTHREADS) Wihb[i] = 0;
    for (int i = tid; i < HP * HP; i += NTHREADS) Whhb[i] = 0;
    for (int i = tid; i < 2 * BM * KP; i += NTHREADS) Xb[0][i] = 0;
    for (int i = tid; i < 2 * BM * HP; i += NTHREADS) Hb[0][i] = 0;

    // ---- stage W_ih [50][272] fp32 -> bf16 LDS [64][288] ----
    for (int p = tid; p < H_SZ * (I_SZ / 4); p += NTHREADS) {
        int n = p / (I_SZ / 4), kq = p - n * (I_SZ / 4);
        float4 v = reinterpret_cast<const float4*>(W_ih)[p];
        *reinterpret_cast<uint2*>(&Wihb[n * KP + kq * 4]) =
            make_uint2(pack2(v.x, v.y), pack2(v.z, v.w));
    }
    // ---- stage W_hh [50][50] -> bf16 LDS [64][64] ----
    for (int p = tid; p < H_SZ * H_SZ; p += NTHREADS) {
        int n = p / H_SZ, k = p - n * H_SZ;
        Whhb[n * HP + k] = bf_hi(W_hh[p]);
    }
    for (int i = tid; i < HP; i += NTHREADS) {
        biasl[i] = (i < H_SZ) ? (b_ih[i] + b_hh[i]) : 0.0f;
        Wfcl[i]  = (i < H_SZ) ? W_fc[i] : 0.0f;
    }

    // ---- staging geometry: slot i -> float4 index p = tid + 128*i in the x tile.
    // Per-slot constant 32-bit BYTE offset (row part); uniform time base advances.
    unsigned soff[NSLOT + 1];
    int sdst[NSLOT + 1];
    #pragma unroll
    for (int i = 0; i <= NSLOT; ++i) {
        int p = tid + NTHREADS * i;
        int m = p / (I_SZ / 4), kq = p - m * (I_SZ / 4);
        if (m >= BM) m = BM - 1;                      // clamp dead tail slot
        soff[i] = (unsigned)(((b0 + m) * T_LEN) * I_SZ + kq * 4) * 4u;  // bytes
        sdst[i] = m * KP + kq * 4;
    }
    // prefetch line offsets (128B lines)
    unsigned pfo0, pfo1;
    {
        int m0 = tid / 9, c0 = tid - (tid / 9) * 9;
        pfo0 = (unsigned)(((b0 + m0) * T_LEN) * I_SZ + c0 * 32) * 4u;
        int p1 = tid + NTHREADS;
        int m1 = p1 / 9, c1 = p1 - m1 * 9;
        if (m1 >= BM) m1 = BM - 1;
        pfo1 = (unsigned)(((b0 + m1) * T_LEN) * I_SZ + c1 * 32) * 4u;
    }
    const char* xbytes = reinterpret_cast<const char*>(x);

#define XLOAD(i, t)  (*reinterpret_cast<const float4*>(xbytes + (size_t)(t) * (I_SZ * 4) + soff[i]))

    // ---- stage t = 0 into buffer 0 ----
    #pragma unroll
    for (int i = 0; i < NSLOT; ++i) {
        float4 v = XLOAD(i, 0);
        *reinterpret_cast<uint2*>(&Xb[0][sdst[i]]) = make_uint2(pack2(v.x, v.y), pack2(v.z, v.w));
    }
    if (tid < TAILQ) {
        float4 v = XLOAD(NSLOT, 0);
        *reinterpret_cast<uint2*>(&Xb[0][sdst[NSLOT]]) = make_uint2(pack2(v.x, v.y), pack2(v.z, v.w));
    }
    __syncthreads();

    // ---- hoist loop-invariant B fragments into VGPRs ----
    v16bf Bih[NKK];
    #pragma unroll
    for (int kk = 0; kk < NKK; ++kk)
        Bih[kk] = load_B32(Wihb, KP, kk * 32, wv * 16 + m16, hi);
    v16bf Bhh[2];
    #pragma unroll
    for (int c = 0; c < 2; ++c)
        Bhh[c] = load_B32(Whhb, HP, c * 32, wv * 16 + m16, hi);

    const float bn = biasl[wv * 16 + m16];
    float ht[8];

// One timestep with compile-time buffer parity
#define RNN_STEP(T_, CUR_, NXT_)                                                        \
    do {                                                                                \
        const int t_ = (T_);                                                            \
        const bool have_next = (t_ + 1 < T_LEN);                                        \
        float4 xs[NSLOT + 1];                                                           \
        if (have_next) {                                                                \
            _Pragma("unroll")                                                           \
            for (int i = 0; i < NSLOT; ++i) xs[i] = XLOAD(i, t_ + 1);                   \
            if (tid < TAILQ) xs[NSLOT] = XLOAD(NSLOT, t_ + 1);                          \
            if (t_ + 2 < T_LEN) {                                                       \
                __builtin_prefetch(xbytes + (size_t)(t_ + 2) * (I_SZ * 4) + pfo0, 0, 1);\
                if (tid < NPFL - NTHREADS)                                              \
                    __builtin_prefetch(xbytes + (size_t)(t_ + 2) * (I_SZ * 4) + pfo1, 0, 1);\
            }                                                                           \
        }                                                                               \
        v16bf aX[NKK], aH[2];                                                           \
        _Pragma("unroll")                                                               \
        for (int kk = 0; kk < NKK; ++kk)                                                \
            aX[kk] = load_A32(Xb[CUR_], KP, kk * 32, m16, hi);                          \
        _Pragma("unroll")                                                               \
        for (int c = 0; c < 2; ++c)                                                     \
            aH[c] = load_A32(Hb[CUR_], HP, c * 32, m16, hi);                            \
        v8f acc0, acc1;                                                                 \
        _Pragma("unroll")                                                               \
        for (int r = 0; r < 8; ++r) { acc0[r] = bn; acc1[r] = 0.0f; }                   \
        _Pragma("unroll")                                                               \
        for (int kk = 0; kk < NKK; ++kk) {                                              \
            if (kk & 1)                                                                 \
                acc1 = __builtin_amdgcn_wmma_f32_16x16x32_bf16(false, aX[kk], false,    \
                           Bih[kk], (short)0, acc1, false, false);                      \
            else                                                                        \
                acc0 = __builtin_amdgcn_wmma_f32_16x16x32_bf16(false, aX[kk], false,    \
                           Bih[kk], (short)0, acc0, false, false);                      \
        }                                                                               \
        acc1 = __builtin_amdgcn_wmma_f32_16x16x32_bf16(false, aH[0], false, Bhh[0],     \
                   (short)0, acc1, false, false);                                       \
        acc0 = __builtin_amdgcn_wmma_f32_16x16x32_bf16(false, aH[1], false, Bhh[1],     \
                   (short)0, acc0, false, false);                                       \
        _Pragma("unroll")                                                               \
        for (int r = 0; r < 8; ++r) ht[r] = TANH_F32(acc0[r] + acc1[r]);                \
        if (have_next) {                                                                \
            _Pragma("unroll")                                                           \
            for (int i = 0; i < NSLOT; ++i)                                             \
                *reinterpret_cast<uint2*>(&Xb[NXT_][sdst[i]]) =                         \
                    make_uint2(pack2(xs[i].x, xs[i].y), pack2(xs[i].z, xs[i].w));       \
            if (tid < TAILQ)                                                            \
                *reinterpret_cast<uint2*>(&Xb[NXT_][sdst[NSLOT]]) =                     \
                    make_uint2(pack2(xs[NSLOT].x, xs[NSLOT].y),                         \
                               pack2(xs[NSLOT].z, xs[NSLOT].w));                        \
        }                                                                               \
        _Pragma("unroll")                                                               \
        for (int r = 0; r < 8; ++r)                                                     \
            Hb[NXT_][(r + 8 * hi) * HP + wv * 16 + m16] = bf_hi(ht[r]);                 \
        __syncthreads();                                                                \
    } while (0)

    for (int t = 0; t < T_LEN; t += 2) {
        RNN_STEP(t,     0, 1);
        RNN_STEP(t + 1, 1, 0);
    }
#undef RNN_STEP
#undef XLOAD

    // ---- head: out[b] = h_T @ W_fc^T + b_fc (keep h in f32) ----
    #pragma unroll
    for (int r = 0; r < 8; ++r)
        Hf[(r + 8 * hi) * HP + wv * 16 + m16] = ht[r];
    __syncthreads();

    if (tid < BM) {
        float s = b_fc[0];
        #pragma unroll 10
        for (int n = 0; n < H_SZ; ++n) s += Hf[tid * HP + n] * Wfcl[n];
        out[b0 + tid] = s;
    }
}

extern "C" void kernel_launch(void* const* d_in, const int* in_sizes, int n_in,
                              void* d_out, int out_size, void* d_ws, size_t ws_size,
                              hipStream_t stream) {
    const float* x    = (const float*)d_in[0];
    const float* W_ih = (const float*)d_in[1];
    const float* W_hh = (const float*)d_in[2];
    const float* b_ih = (const float*)d_in[3];
    const float* b_hh = (const float*)d_in[4];
    const float* W_fc = (const float*)d_in[5];
    const float* b_fc = (const float*)d_in[6];
    float* out = (float*)d_out;

    dim3 grid(B_TOT / BM);
    dim3 block(NTHREADS);
    hipLaunchKernelGGL(rnn_fused_kernel, grid, block, 0, stream,
                       x, W_ih, W_hh, b_ih, b_hh, W_fc, b_fc, out);
}
